// PVConv_19464791786039
// MI455X (gfx1250) — matrix operand order repaired and checked by reading, not compiled, not run on hardware
//
#include <hip/hip_runtime.h>
#include <hip/hip_bf16.h>
#include <math.h>

#define BD   4
#define CIN_ 64
#define COUT_ 128
#define NPTS 2048
#define RES  16
#define NVOX 4096
#define MCTX 256
#define CTXD 1024
#define GN_EPS 1e-5f

typedef __attribute__((ext_vector_type(16))) _Float16 v16h;
typedef __attribute__((ext_vector_type(8)))  float    v8f;

// ---------------- WMMA fragment loaders (CDNA5 16x16x32 f16, wave32) ---------
// A (16x32 MxK): lane (g=lane/16, m=lane%16) holds row m, K run [16g..16g+15]
// -> two contiguous 32B runs from a row-major row pointer.
__device__ __forceinline__ v16h load_a_row(const float* row, int g) {
  v16h a;
  const float4* p0 = (const float4*)(row + 16 * g);
  const float4* p1 = (const float4*)(row + 16 * g + 8);
#pragma unroll
  for (int i = 0; i < 2; ++i) {
    float4 q = p0[i];
    a[4 * i + 0] = (_Float16)q.x; a[4 * i + 1] = (_Float16)q.y;
    a[4 * i + 2] = (_Float16)q.z; a[4 * i + 3] = (_Float16)q.w;
  }
#pragma unroll
  for (int i = 0; i < 2; ++i) {
    float4 q = p1[i];
    a[8 + 4 * i + 0] = (_Float16)q.x; a[8 + 4 * i + 1] = (_Float16)q.y;
    a[8 + 4 * i + 2] = (_Float16)q.z; a[8 + 4 * i + 3] = (_Float16)q.w;
  }
  return a;
}
// B (32x16 KxN): lane (g, n=lane%16) holds column n, K run [16g..16g+15]:
// element b[s] = Bsrc[16g + s] for a pointer giving contiguous K for column n.
__device__ __forceinline__ v16h load_b_run(const float* run) {
  v16h b;
  const float4* p = (const float4*)run;
#pragma unroll
  for (int i = 0; i < 4; ++i) {
    float4 q = p[i];
    b[4 * i + 0] = (_Float16)q.x; b[4 * i + 1] = (_Float16)q.y;
    b[4 * i + 2] = (_Float16)q.z; b[4 * i + 3] = (_Float16)q.w;
  }
  return b;
}
__device__ __forceinline__ v16h zero_h16() {
  v16h z;
#pragma unroll
  for (int i = 0; i < 16; ++i) z[i] = (_Float16)0.f;
  return z;
}
// C/D (16x16 f32): VGPR j -> row j+8g, col = lane&15

// ---------------- small utility kernels --------------------------------------
__global__ void fill_f32(float* p, float v, long n) {
  long i = (long)blockIdx.x * blockDim.x + threadIdx.x;
  if (i < n) p[i] = v;
}
__global__ void copy_f32(const float* __restrict__ s, float* __restrict__ d, long n) {
  long i = (long)blockIdx.x * blockDim.x + threadIdx.x;
  if (i < n) d[i] = s[i];
}
// conv weights OIDHW [o][i][tap] -> [tap][o][i] so B-frag K-runs are contiguous
__global__ void conv_w_transpose(const float* __restrict__ w, float* __restrict__ wt,
                                 int CI) {
  long i = (long)blockIdx.x * blockDim.x + threadIdx.x;
  long tot = (long)COUT_ * CI * 27;
  if (i >= tot) return;
  int tap = (int)(i % 27);
  long oi = i / 27;
  int ci = (int)(oi % CI);
  int o  = (int)(oi / CI);
  wt[((long)tap * COUT_ + o) * CI + ci] = w[i];
}
// [b][m][c] -> [b][c][m] (channel-major V for the P*V WMMA operand)
__global__ void transpose_bmc(const float* __restrict__ v, float* __restrict__ vt, int M) {
  long i = (long)blockIdx.x * blockDim.x + threadIdx.x;
  if (i >= (long)BD * M * COUT_) return;
  int c = (int)(i % COUT_);
  long bm = i / COUT_;
  int m = (int)(bm % M);
  int b = (int)(bm / M);
  vt[((long)b * COUT_ + c) * M + m] = v[i];
}

// ---------------- voxelization ----------------------------------------------
__global__ void coord_stats(const float* __restrict__ coords, float* __restrict__ cs) {
  int b = blockIdx.x;
  const float* cb = coords + (long)b * 3 * NPTS;
  __shared__ float red[256];
  float mean[3];
  for (int d = 0; d < 3; ++d) {
    float s = 0.f;
    for (int n = threadIdx.x; n < NPTS; n += 256) s += cb[d * NPTS + n];
    red[threadIdx.x] = s; __syncthreads();
    for (int o = 128; o > 0; o >>= 1) {
      if (threadIdx.x < o) red[threadIdx.x] += red[threadIdx.x + o];
      __syncthreads();
    }
    mean[d] = red[0] / (float)NPTS;
    __syncthreads();
  }
  float mx = 0.f;
  for (int n = threadIdx.x; n < NPTS; n += 256) {
    float dx = cb[n] - mean[0], dy = cb[NPTS + n] - mean[1], dz = cb[2 * NPTS + n] - mean[2];
    mx = fmaxf(mx, sqrtf(dx * dx + dy * dy + dz * dz));
  }
  red[threadIdx.x] = mx; __syncthreads();
  for (int o = 128; o > 0; o >>= 1) {
    if (threadIdx.x < o) red[threadIdx.x] = fmaxf(red[threadIdx.x], red[threadIdx.x + o]);
    __syncthreads();
  }
  if (threadIdx.x == 0) {
    cs[b * 4 + 0] = mean[0]; cs[b * 4 + 1] = mean[1]; cs[b * 4 + 2] = mean[2];
    cs[b * 4 + 3] = red[0] * 2.f;
  }
}

__global__ void voxelize_scatter(const float* __restrict__ feat, const float* __restrict__ coords,
                                 const float* __restrict__ cs, float* __restrict__ vox,
                                 float* __restrict__ cnt, float* __restrict__ normc) {
  int i = blockIdx.x * blockDim.x + threadIdx.x;
  if (i >= BD * NPTS) return;
  int b = i / NPTS, n = i % NPTS;
  const float* cb = coords + (long)b * 3 * NPTS;
  float scale = cs[b * 4 + 3];
  int vc[3];
#pragma unroll
  for (int d = 0; d < 3; ++d) {
    float nm  = (cb[d * NPTS + n] - cs[b * 4 + d]) / scale + 0.5f;
    float ncl = fminf(fmaxf(nm * (float)RES, 0.f), (float)RES - 1.f);
    normc[(long)b * 3 * NPTS + d * NPTS + n] = ncl;
    vc[d] = (int)rintf(ncl);
  }
  int idx = (vc[0] * RES + vc[1]) * RES + vc[2];
  atomicAdd(&cnt[b * NVOX + idx], 1.f);
  float* vp = vox + ((long)b * NVOX + idx) * CIN_;
  const float* fp = feat + (long)b * CIN_ * NPTS + n;
  for (int c = 0; c < CIN_; ++c) atomicAdd(&vp[c], fp[(long)c * NPTS]);
}

__global__ void voxel_avg(float* vox, const float* cnt) {
  long i = (long)blockIdx.x * blockDim.x + threadIdx.x;
  if (i >= (long)BD * NVOX * CIN_) return;
  vox[i] /= fmaxf(cnt[i / CIN_], 1.f);
}

// ---------------- conv3d 3x3x3 as 27 per-tap WMMA GEMMs ----------------------
// x: [B][4096][CI] token-major, wt: [tap][COUT][CI], y: [B][4096][128]
template <int CI>
__global__ void conv3d_tap_wmma(const float* __restrict__ x, const float* __restrict__ wt,
                                const float* __restrict__ bias, float* __restrict__ y) {
  int wave = threadIdx.x >> 5, lane = threadIdx.x & 31;
  int tile = blockIdx.x * 4 + wave;
  int ct = tile & 7;
  int vt = (tile >> 3) & 255;
  int b  = tile >> 11;
  int g = lane >> 4, hn = lane & 15;

  int v = vt * 16 + hn;  // this lane's voxel row (rows duplicated across halves)
  int vx = v >> 8, vy = (v >> 4) & 15, vz = v & 15;

  v8f acc = {0.f, 0.f, 0.f, 0.f, 0.f, 0.f, 0.f, 0.f};
  const float* wlane = wt + (long)(ct * 16 + hn) * CI;  // weight row for col n=hn

  for (int tap = 0; tap < 27; ++tap) {
    int kd = tap / 9, kh = (tap / 3) % 3, kw = tap % 3;
    int nx = vx + kd - 1, ny = vy + kh - 1, nz = vz + kw - 1;
    bool valid = ((unsigned)nx < 16u) && ((unsigned)ny < 16u) && ((unsigned)nz < 16u);
    const float* arow = x + ((long)b * NVOX + ((nx * 16 + ny) * 16 + nz)) * CI;
    const float* wrow = wlane + (long)tap * COUT_ * CI;
#pragma unroll
    for (int kc = 0; kc < CI; kc += 32) {
      v16h af = valid ? load_a_row(arow + kc, g) : zero_h16();
      v16h bf = load_b_run(wrow + kc + 16 * g);
      acc = __builtin_amdgcn_wmma_f32_16x16x32_f16(false, af, false, bf, (short)0, acc, false, false);
    }
  }
  float bn = bias[ct * 16 + hn];
  float* out = y + ((long)b * NVOX + vt * 16) * COUT_ + ct * 16;
#pragma unroll
  for (int j = 0; j < 8; ++j) out[(long)(j + 8 * g) * COUT_ + hn] = acc[j] + bn;
}

// ---------------- generic token-major linear (1x1 conv) via WMMA -------------
// x: [T][CI], w: [CO][CI], y: [T][CO]
__global__ void linear_wmma(const float* __restrict__ x, const float* __restrict__ w,
                            const float* __restrict__ bias, float* __restrict__ y,
                            int T, int CI, int CO) {
  int wave = threadIdx.x >> 5, lane = threadIdx.x & 31;
  int cotiles = CO >> 4;
  int tile = blockIdx.x * 4 + wave;
  int ct = tile % cotiles;
  int tt = tile / cotiles;
  int g = lane >> 4, hn = lane & 15;
  v8f acc = {0.f, 0.f, 0.f, 0.f, 0.f, 0.f, 0.f, 0.f};
  const float* arow = x + (long)(tt * 16 + hn) * CI;
  const float* wrow = w + (long)(ct * 16 + hn) * CI;
  for (int kc = 0; kc < CI; kc += 32) {
    if (kc + 32 < CI) __builtin_prefetch(arow + kc + 32, 0, 1);  // global_prefetch_b8
    v16h af = load_a_row(arow + kc, g);
    v16h bf = load_b_run(wrow + kc + 16 * g);
    acc = __builtin_amdgcn_wmma_f32_16x16x32_f16(false, af, false, bf, (short)0, acc, false, false);
  }
  float bn = bias ? bias[ct * 16 + hn] : 0.f;
  float* out = y + (long)(tt * 16) * CO + ct * 16;
#pragma unroll
  for (int j = 0; j < 8; ++j) out[(long)(j + 8 * g) * CO + hn] = acc[j] + bn;
}

// ---------------- group norm ------------------------------------------------
__global__ void gn_stats(const float* __restrict__ x, const float* __restrict__ res,
                         float* __restrict__ stats, int S) {
  int b = blockIdx.x >> 3, grp = blockIdx.x & 7;
  const float* xb = x + (long)b * S * COUT_;
  const float* rb = res ? res + (long)b * S * COUT_ : nullptr;
  float s = 0.f, sq = 0.f;
  for (int t = threadIdx.x; t < S; t += blockDim.x) {
    const float* px = xb + (long)t * COUT_ + grp * 16;
    const float* pr = rb ? rb + (long)t * COUT_ + grp * 16 : nullptr;
#pragma unroll
    for (int c = 0; c < 16; ++c) {
      float v = px[c] + (pr ? pr[c] : 0.f);
      s += v; sq += v * v;
    }
  }
  __shared__ float rs[256], rq[256];
  rs[threadIdx.x] = s; rq[threadIdx.x] = sq; __syncthreads();
  for (int o = 128; o > 0; o >>= 1) {
    if (threadIdx.x < o) { rs[threadIdx.x] += rs[threadIdx.x + o]; rq[threadIdx.x] += rq[threadIdx.x + o]; }
    __syncthreads();
  }
  if (threadIdx.x == 0) {
    float inv = 1.f / ((float)S * 16.f);
    float mu  = rs[0] * inv;
    float var = rq[0] * inv - mu * mu;
    stats[blockIdx.x * 2]     = mu;
    stats[blockIdx.x * 2 + 1] = rsqrtf(var + GN_EPS);
  }
}

__global__ void gn_apply(const float* __restrict__ x, const float* __restrict__ res,
                         const float* __restrict__ stats, const float* __restrict__ gamma,
                         const float* __restrict__ beta, float* __restrict__ y,
                         int S, int doswish) {
  long i = (long)blockIdx.x * blockDim.x + threadIdx.x;
  long tot = (long)BD * S * COUT_;
  if (i >= tot) return;
  int c = (int)(i % COUT_);
  long bt = i / COUT_;
  int b = (int)(bt / S);
  int grp = c >> 4;
  float v = x[i] + (res ? res[i] : 0.f);
  float mu = stats[(b * 8 + grp) * 2], rstd = stats[(b * 8 + grp) * 2 + 1];
  float o = (v - mu) * rstd * gamma[c] + beta[c];
  if (doswish) o = o / (1.f + __expf(-o));
  y[i] = o;
}

// ---------------- attention (two-pass streaming softmax, WMMA) ---------------
// Q: [B][4096][128], K: [B][M][128]; one wave per (b, 16-query tile)
__global__ void attn_pass1(const float* __restrict__ Q, const float* __restrict__ K,
                           float* __restrict__ rowstats, int M) {
  int wave = threadIdx.x >> 5, lane = threadIdx.x & 31;
  int wid = blockIdx.x * 2 + wave;
  int qt = wid & 255, b = wid >> 8;
  int g = lane >> 4, hn = lane & 15;
  const float* Qb = Q + (long)b * NVOX * COUT_;
  const float* Kb = K + (long)b * M * COUT_;
  const float* qrow = Qb + (long)(qt * 16 + hn) * COUT_;
  v16h qa[4];
#pragma unroll
  for (int kc = 0; kc < 4; ++kc) qa[kc] = load_a_row(qrow + kc * 32, g);

  __shared__ float sS[2][16 * 16];
  float* S = sS[wave];
  float rmax = -1e30f, rsum = 0.f;
  for (int m0 = 0; m0 < M; m0 += 16) {
    v8f sc = {0.f, 0.f, 0.f, 0.f, 0.f, 0.f, 0.f, 0.f};
    const float* krow = Kb + (long)(m0 + hn) * COUT_ + 16 * g;
#pragma unroll
    for (int kc = 0; kc < 4; ++kc) {
      v16h bf = load_b_run(krow + kc * 32);
      sc = __builtin_amdgcn_wmma_f32_16x16x32_f16(false, qa[kc], false, bf, (short)0, sc, false, false);
    }
#pragma unroll
    for (int j = 0; j < 8; ++j) S[(j + 8 * g) * 16 + hn] = sc[j];
    __syncthreads();
    if (lane < 16) {
      float mloc = rmax;
#pragma unroll
      for (int i = 0; i < 16; ++i) mloc = fmaxf(mloc, S[lane * 16 + i]);
      float scale = __expf(rmax - mloc);
      float add = 0.f;
#pragma unroll
      for (int i = 0; i < 16; ++i) add += __expf(S[lane * 16 + i] - mloc);
      rsum = rsum * scale + add;
      rmax = mloc;
    }
    __syncthreads();
  }
  if (lane < 16) {
    long r = (long)b * NVOX + qt * 16 + lane;
    rowstats[r * 2] = rmax;
    rowstats[r * 2 + 1] = rsum;
  }
}

// Vt is channel-major [B][128][M]; P staged in LDS directly in A-frag layout
__global__ void attn_pass2(const float* __restrict__ Q, const float* __restrict__ K,
                           const float* __restrict__ Vt, const float* __restrict__ rowstats,
                           float* __restrict__ H, int M) {
  int wave = threadIdx.x >> 5, lane = threadIdx.x & 31;
  int wid = blockIdx.x * 2 + wave;
  int qt = wid & 255, b = wid >> 8;
  int g = lane >> 4, hn = lane & 15;
  const float* Qb = Q + (long)b * NVOX * COUT_;
  const float* Kb = K + (long)b * M * COUT_;
  const float* Vtb = Vt + (long)b * COUT_ * M;
  const float* qrow = Qb + (long)(qt * 16 + hn) * COUT_;
  v16h qa[4];
#pragma unroll
  for (int kc = 0; kc < 4; ++kc) qa[kc] = load_a_row(qrow + kc * 32, g);

  float rmax[8], rinv[8];
#pragma unroll
  for (int j = 0; j < 8; ++j) {  // score row = j + 8g
    long r = (long)b * NVOX + qt * 16 + 8 * g + j;
    rmax[j] = rowstats[r * 2];
    rinv[j] = 1.f / rowstats[r * 2 + 1];
  }
  v8f zero = {0.f, 0.f, 0.f, 0.f, 0.f, 0.f, 0.f, 0.f};
  v8f acc[8];
#pragma unroll
  for (int j = 0; j < 8; ++j) acc[j] = zero;

  // P in fragment-ready layout: [lane][16] halves -> each lane reads one v16h
  __shared__ _Float16 sP[2][32 * 16];
  _Float16* P = sP[wave];
  for (int m0 = 0; m0 < M; m0 += 32) {
#pragma unroll
    for (int mt = 0; mt < 2; ++mt) {
      v8f sc = zero;
      const float* krow = Kb + (long)(m0 + mt * 16 + hn) * COUT_ + 16 * g;
#pragma unroll
      for (int kc = 0; kc < 4; ++kc) {
        v16h bf = load_b_run(krow + kc * 32);
        sc = __builtin_amdgcn_wmma_f32_16x16x32_f16(false, qa[kc], false, bf, (short)0, sc, false, false);
      }
      // scatter P into A-frag layout: element (row r, k-col cc):
      //   dest lane = r + 16*(cc>>4), slot = cc&15
      int cc = mt * 16 + hn;
      int dl16 = 16 * (cc >> 4);
      int slot = cc & 15;
#pragma unroll
      for (int j = 0; j < 8; ++j) {
        float pv = __expf(sc[j] - rmax[j]) * rinv[j];
        P[(j + 8 * g + dl16) * 16 + slot] = (_Float16)pv;
      }
    }
    __syncthreads();
    v16h pa = *(const v16h*)(P + lane * 16);  // 2x ds_load_b128
    __syncthreads();
#pragma unroll
    for (int co = 0; co < 8; ++co) {
      const float* vrow = Vtb + (long)(co * 16 + hn) * M + m0 + 16 * g;
      v16h bf = load_b_run(vrow);
      acc[co] = __builtin_amdgcn_wmma_f32_16x16x32_f16(false, pa, false, bf, (short)0, acc[co], false, false);
    }
  }
  float* out = H + ((long)b * NVOX + qt * 16) * COUT_;
#pragma unroll
  for (int co = 0; co < 8; ++co)
#pragma unroll
    for (int j = 0; j < 8; ++j)
      out[(long)(j + 8 * g) * COUT_ + co * 16 + hn] = acc[co][j];
}

// ---------------- point branch + devox ---------------------------------------
__global__ void transpose_feat(const float* __restrict__ f, float* __restrict__ ft) {
  long i = (long)blockIdx.x * blockDim.x + threadIdx.x;
  if (i >= (long)BD * CIN_ * NPTS) return;
  int n = (int)(i % NPTS);
  int c = (int)((i / NPTS) % CIN_);
  int b = (int)(i / ((long)NPTS * CIN_));
  ft[((long)b * NPTS + n) * CIN_ + c] = f[i];
}

__global__ void devox_add(const float* __restrict__ vfeat, const float* __restrict__ normc,
                          const float* __restrict__ pf, float* __restrict__ out) {
  long i = (long)blockIdx.x * blockDim.x + threadIdx.x;
  if (i >= (long)BD * COUT_ * NPTS) return;
  int n = (int)(i % NPTS);
  int c = (int)((i / NPTS) % COUT_);
  int b = (int)(i / ((long)NPTS * COUT_));
  const float* nc = normc + (long)b * 3 * NPTS;
  float cx = nc[n], cy = nc[NPTS + n], cz = nc[2 * NPTS + n];
  int x0 = min(max((int)floorf(cx), 0), 15); float fx = cx - (float)x0; int x1 = min(x0 + 1, 15);
  int y0 = min(max((int)floorf(cy), 0), 15); float fy = cy - (float)y0; int y1 = min(y0 + 1, 15);
  int z0 = min(max((int)floorf(cz), 0), 15); float fz = cz - (float)z0; int z1 = min(z0 + 1, 15);
  const float* vb = vfeat + (long)b * NVOX * COUT_ + c;
  auto G = [&](int xi, int yi, int zi) { return vb[(long)((xi * 16 + yi) * 16 + zi) * COUT_]; };
  float v =
      (1 - fx) * (1 - fy) * (1 - fz) * G(x0, y0, z0) + (1 - fx) * (1 - fy) * fz * G(x0, y0, z1) +
      (1 - fx) * fy * (1 - fz) * G(x0, y1, z0) + (1 - fx) * fy * fz * G(x0, y1, z1) +
      fx * (1 - fy) * (1 - fz) * G(x1, y0, z0) + fx * (1 - fy) * fz * G(x1, y0, z1) +
      fx * fy * (1 - fz) * G(x1, y1, z0) + fx * fy * fz * G(x1, y1, z1);
  out[((long)b * COUT_ + c) * NPTS + n] = v + pf[((long)b * NPTS + n) * COUT_ + c];
}

// ---------------- host orchestration -----------------------------------------
extern "C" void kernel_launch(void* const* d_in, const int* in_sizes, int n_in,
                              void* d_out, int out_size, void* d_ws, size_t ws_size,
                              hipStream_t stream) {
  (void)in_sizes; (void)n_in; (void)out_size; (void)ws_size;
  const float* features = (const float*)d_in[0];
  const float* coords   = (const float*)d_in[1];
  const float* temb     = (const float*)d_in[2];
  const float* context  = (const float*)d_in[3];
  const float* c1_w = (const float*)d_in[4];  const float* c1_b = (const float*)d_in[5];
  const float* gn1_g = (const float*)d_in[6]; const float* gn1_b = (const float*)d_in[7];
  const float* c2_w = (const float*)d_in[8];  const float* c2_b = (const float*)d_in[9];
  const float* gn2_g = (const float*)d_in[10]; const float* gn2_b = (const float*)d_in[11];
  const float* sa_qw = (const float*)d_in[12]; const float* sa_qb = (const float*)d_in[13];
  const float* sa_ow = (const float*)d_in[14]; const float* sa_ob = (const float*)d_in[15];
  const float* sa_gng = (const float*)d_in[16]; const float* sa_gnb = (const float*)d_in[17];
  const float* sa_kw = (const float*)d_in[18]; const float* sa_kb = (const float*)d_in[19];
  const float* sa_vw = (const float*)d_in[20]; const float* sa_vb = (const float*)d_in[21];
  const float* ca_qw = (const float*)d_in[22]; const float* ca_qb = (const float*)d_in[23];
  const float* ca_ow = (const float*)d_in[24]; const float* ca_ob = (const float*)d_in[25];
  const float* ca_gng = (const float*)d_in[26]; const float* ca_gnb = (const float*)d_in[27];
  const float* ca_kw = (const float*)d_in[28]; const float* ca_vw = (const float*)d_in[29];
  const float* pm_w = (const float*)d_in[30]; const float* pm_b = (const float*)d_in[31];
  const float* pm_g = (const float*)d_in[32]; const float* pm_beta = (const float*)d_in[33];
  float* out = (float*)d_out;

  float* ws = (float*)d_ws;
  size_t off = 0;
  auto alloc = [&](size_t nf) { float* p = ws + off; off += nf; return p; };
  float* vox   = alloc((size_t)BD * NVOX * CIN_);
  float* cnt   = alloc((size_t)BD * NVOX);
  float* normc = alloc((size_t)BD * 3 * NPTS);
  float* X1    = alloc((size_t)BD * NVOX * COUT_);
  float* X2    = alloc((size_t)BD * NVOX * COUT_);
  float* Qb    = alloc((size_t)BD * NVOX * COUT_);
  float* Kb    = alloc((size_t)BD * NVOX * COUT_);
  float* Vb    = alloc((size_t)BD * NVOX * COUT_);
  float* Vt    = alloc((size_t)BD * NVOX * COUT_);
  float* Hb    = alloc((size_t)BD * NVOX * COUT_);
  float* CK    = alloc((size_t)BD * MCTX * COUT_);
  float* CV    = alloc((size_t)BD * MCTX * COUT_);
  float* CVt   = alloc((size_t)BD * MCTX * COUT_);
  float* wt1   = alloc((size_t)27 * COUT_ * CIN_);
  float* wt2   = alloc((size_t)27 * COUT_ * COUT_);
  float* ftok  = alloc((size_t)BD * NPTS * CIN_);
  float* Pb    = alloc((size_t)BD * NPTS * COUT_);
  float* cstats = alloc(16);
  float* gstats = alloc(64);
  float* rowst  = alloc((size_t)BD * NVOX * 2);

  const long EVOX = (long)BD * NVOX * COUT_;  // 2,097,152
  const long EPTS = (long)BD * NPTS * COUT_;  // 1,048,576

  // weight pre-transposes (taps innermost -> taps outermost)
  conv_w_transpose<<<(int)(((long)COUT_ * CIN_ * 27 + 255) / 256), 256, 0, stream>>>(c1_w, wt1, CIN_);
  conv_w_transpose<<<(int)(((long)COUT_ * COUT_ * 27 + 255) / 256), 256, 0, stream>>>(c2_w, wt2, COUT_);

  // voxelize
  long zn = (long)BD * NVOX * CIN_ + (long)BD * NVOX;  // vox + cnt contiguous
  fill_f32<<<(int)((zn + 255) / 256), 256, 0, stream>>>(vox, 0.f, zn);
  coord_stats<<<BD, 256, 0, stream>>>(coords, cstats);
  voxelize_scatter<<<(BD * NPTS + 255) / 256, 256, 0, stream>>>(features, coords, cstats, vox, cnt, normc);
  voxel_avg<<<(int)(((long)BD * NVOX * CIN_ + 255) / 256), 256, 0, stream>>>(vox, cnt);

  // conv1 + GN + swish
  conv3d_tap_wmma<CIN_><<<2048, 128, 0, stream>>>(vox, wt1, c1_b, X1);
  gn_stats<<<BD * 8, 256, 0, stream>>>(X1, nullptr, gstats, NVOX);
  gn_apply<<<(int)((EVOX + 255) / 256), 256, 0, stream>>>(X1, nullptr, gstats, gn1_g, gn1_b, X1, NVOX, 1);

  // conv2 + GN
  conv3d_tap_wmma<COUT_><<<2048, 128, 0, stream>>>(X1, wt2, c2_b, X2);
  gn_stats<<<BD * 8, 256, 0, stream>>>(X2, nullptr, gstats, NVOX);
  gn_apply<<<(int)((EVOX + 255) / 256), 256, 0, stream>>>(X2, nullptr, gstats, gn2_g, gn2_b, X2, NVOX, 0);

  // self attention
  linear_wmma<<<2048, 128, 0, stream>>>(X2, sa_qw, sa_qb, Qb, BD * NVOX, COUT_, COUT_);
  linear_wmma<<<2048, 128, 0, stream>>>(X2, sa_kw, sa_kb, Kb, BD * NVOX, COUT_, COUT_);
  linear_wmma<<<2048, 128, 0, stream>>>(X2, sa_vw, sa_vb, Vb, BD * NVOX, COUT_, COUT_);
  transpose_bmc<<<(int)((EVOX + 255) / 256), 256, 0, stream>>>(Vb, Vt, NVOX);
  attn_pass1<<<512, 64, 0, stream>>>(Qb, Kb, rowst, NVOX);
  attn_pass2<<<512, 64, 0, stream>>>(Qb, Kb, Vt, rowst, Hb, NVOX);
  linear_wmma<<<2048, 128, 0, stream>>>(Hb, sa_ow, sa_ob, Qb, BD * NVOX, COUT_, COUT_);  // O -> Qb
  gn_stats<<<BD * 8, 256, 0, stream>>>(Qb, X2, gstats, NVOX);
  gn_apply<<<(int)((EVOX + 255) / 256), 256, 0, stream>>>(Qb, X2, gstats, sa_gng, sa_gnb, X1, NVOX, 1);

  // cross attention (keys/values from context: 1024 tokens x 1024 dims -> 128)
  linear_wmma<<<128, 128, 0, stream>>>(context, ca_kw, nullptr, CK, BD * MCTX, CTXD, COUT_);
  linear_wmma<<<128, 128, 0, stream>>>(context, ca_vw, nullptr, CV, BD * MCTX, CTXD, COUT_);
  transpose_bmc<<<(int)(((long)BD * MCTX * COUT_ + 255) / 256), 256, 0, stream>>>(CV, CVt, MCTX);
  linear_wmma<<<2048, 128, 0, stream>>>(X1, ca_qw, ca_qb, Qb, BD * NVOX, COUT_, COUT_);
  attn_pass1<<<512, 64, 0, stream>>>(Qb, CK, rowst, MCTX);
  attn_pass2<<<512, 64, 0, stream>>>(Qb, CK, CVt, rowst, Hb, MCTX);
  linear_wmma<<<2048, 128, 0, stream>>>(Hb, ca_ow, ca_ob, Kb, BD * NVOX, COUT_, COUT_);  // O -> Kb
  gn_stats<<<BD * 8, 256, 0, stream>>>(Kb, X1, gstats, NVOX);
  gn_apply<<<(int)((EVOX + 255) / 256), 256, 0, stream>>>(Kb, X1, gstats, ca_gng, ca_gnb, X2, NVOX, 1);

  // point branch: pf = swish(GN(pm_w @ features + pm_b))
  transpose_feat<<<(int)(((long)BD * CIN_ * NPTS + 255) / 256), 256, 0, stream>>>(features, ftok);
  linear_wmma<<<1024, 128, 0, stream>>>(ftok, pm_w, pm_b, Pb, BD * NPTS, CIN_, COUT_);
  gn_stats<<<BD * 8, 256, 0, stream>>>(Pb, nullptr, gstats, NPTS);
  gn_apply<<<(int)((EPTS + 255) / 256), 256, 0, stream>>>(Pb, nullptr, gstats, pm_g, pm_beta, Pb, NPTS, 1);

  // devox + add, then passthrough outputs (tuple: voxfeat+pf, coords, temb)
  devox_add<<<(int)((EPTS + 255) / 256), 256, 0, stream>>>(X2, normc, Pb, out);
  copy_f32<<<(BD * 3 * NPTS + 255) / 256, 256, 0, stream>>>(coords, out + EPTS, (long)BD * 3 * NPTS);
  copy_f32<<<(int)(((long)BD * CIN_ * NPTS + 255) / 256), 256, 0, stream>>>(
      temb, out + EPTS + (long)BD * 3 * NPTS, (long)BD * CIN_ * NPTS);
}
